// EvolveGCNO_47459388620812
// MI455X (gfx1250) — compile-verified, matrix operand order checked by reading, not run on hardware
//
#include <hip/hip_runtime.h>
#include <hip/hip_bf16.h>

// ---------------------------------------------------------------------------
// EvolveGCN-O forward for MI455X (gfx1250, wave32, WMMA)
//   N=100000 nodes, E=1.6M edges, D=256 channels
//   GEMMs -> bf16 WMMA (v_wmma_f32_16x16x32_bf16), fp32 accumulate
//     - X / h pre-converted to bf16 once (streaming pass), so the GEMM inner
//       loop is pure ds_load_b128 + v_wmma (no conversion VALU)
//     - A tile staged to LDS via gfx1250 async-to-LDS (ASYNCcnt) with sync
//       fallback; B fragments preloaded into registers, reused over 5 stripes
//   propagate -> float4 gather + global_atomic_add_f32 scatter (xw L2-resident)
// ---------------------------------------------------------------------------

#define USE_ASYNC_LDS 1   // gfx1250 global_load_async_to_lds_b128 + s_wait_asynccnt

typedef __attribute__((ext_vector_type(16))) __bf16 v16bf;
typedef __attribute__((ext_vector_type(8)))  float  v8f;

#define DCH 256
#define STRIPES 5          // 16-row stripes per GEMM block; N/16 % STRIPES == 0

struct Q2 { uint4 a, b; }; // 32B carrier for bit-casting into v16bf

// round-to-nearest-even float -> bf16 (pure integer ops; no cast dependency)
__device__ __forceinline__ __bf16 f2bf(float f) {
  unsigned u = __float_as_uint(f);
  unsigned short s = (unsigned short)((u + 0x7FFFu + ((u >> 16) & 1u)) >> 16);
  return __builtin_bit_cast(__bf16, s);
}

__global__ void k_fill(float* __restrict__ p, size_t n, float v) {
  size_t i = (size_t)blockIdx.x * blockDim.x + threadIdx.x;
  if (i < n) p[i] = v;
}

// float[8] -> bf16[8] per thread (RNE), 16B stores; memory-bound streaming pass
__global__ void k_cvt(const float* __restrict__ in, __bf16* __restrict__ out, size_t n8) {
  size_t i = (size_t)blockIdx.x * blockDim.x + threadIdx.x;
  if (i >= n8) return;
  const float4* p = (const float4*)(in + i * 8);
  float4 a = p[0], b = p[1];
  union { __bf16 h[8]; uint4 q; } u;
  u.h[0] = f2bf(a.x); u.h[1] = f2bf(a.y); u.h[2] = f2bf(a.z); u.h[3] = f2bf(a.w);
  u.h[4] = f2bf(b.x); u.h[5] = f2bf(b.y); u.h[6] = f2bf(b.z); u.h[7] = f2bf(b.w);
  *(uint4*)(out + i * 8) = u.q;
}

__global__ void k_deg(const int* __restrict__ dst, float* __restrict__ deg, int E) {
  int e = blockIdx.x * blockDim.x + threadIdx.x;
  if (e < E) atomicAdd(&deg[dst[e]], 1.0f);
}

__global__ void k_dinv(float* __restrict__ d, int n) {
  int i = blockIdx.x * blockDim.x + threadIdx.x;
  if (i < n) { float v = d[i]; d[i] = (v > 0.f) ? rsqrtf(v) : 0.f; }
}

__global__ void k_norm(const int* __restrict__ src, const int* __restrict__ dst,
                       const float* __restrict__ dinv, float* __restrict__ norm, int E) {
  int e = blockIdx.x * blockDim.x + threadIdx.x;
  if (e < E) norm[e] = dinv[src[e]] * dinv[dst[e]];
}

// GRU weight evolution: W = GRU(x=iw, h=iw); writes W^T in bf16 (for B-fragment loads)
__global__ __launch_bounds__(256) void k_gruw(const float* __restrict__ iw,
    const float* __restrict__ wih, const float* __restrict__ whh,
    const float* __restrict__ bih, const float* __restrict__ bhh,
    __bf16* __restrict__ Wt) {
  const int a = blockIdx.x;      // row of iw
  const int j = threadIdx.x;     // output column
  const float* xr = iw + (size_t)a * DCH;
  float gi[3], gh[3];
#pragma unroll
  for (int g = 0; g < 3; ++g) {
    const float* wi = wih + (size_t)(g * DCH + j) * DCH;
    const float* wh = whh + (size_t)(g * DCH + j) * DCH;
    float si = bih[g * DCH + j], sh = bhh[g * DCH + j];
    for (int k = 0; k < DCH; ++k) {
      float x = xr[k];
      si = fmaf(x, wi[k], si);
      sh = fmaf(x, wh[k], sh);
    }
    gi[g] = si; gh[g] = sh;
  }
  float r  = 1.f / (1.f + __expf(-(gi[0] + gh[0])));
  float z  = 1.f / (1.f + __expf(-(gi[1] + gh[1])));
  float nn = tanhf(gi[2] + r * gh[2]);
  float w  = (1.f - z) * nn + z * xr[j];
  Wt[(size_t)j * DCH + a] = f2bf(w);   // transposed: Wt[n][k] = W[k][n]
}

// C[N x 256] = A[N x 256] @ W[256 x 256]; A in bf16, W transposed in bf16.
// Block: 512 thr = 16 waves; processes STRIPES 16-row stripes. Wave w owns
// columns [16w,16w+16): its 8 B fragments are loaded once into registers and
// reused across stripes. Per stripe: stage 8KB A tile in LDS, then 8x
// (2x ds_load_b128 + v_wmma_f32_16x16x32_bf16).
__global__ __launch_bounds__(512) void k_gemm(const __bf16* __restrict__ Ab,
    const __bf16* __restrict__ Wt, float* __restrict__ C, int nstripe) {
  __shared__ __bf16 As[16 * DCH];           // 8 KB
  const int tid  = threadIdx.x;
  const int lane = tid & 31;
  const int wave = tid >> 5;
  const int mrow = lane & 15;               // A row / B,C column index within tile
  const int koff = (lane >> 4) << 3;        // 0 or 8 (half-wave K offset per ISA layout)
  const int ncol = (wave << 4) + mrow;      // global output column

  // preload this wave's 8 B fragments (16 cols x 256 K) into registers
  v16bf bfr[8];
#pragma unroll
  for (int kb = 0; kb < 8; ++kb) {
    const __bf16* bp = Wt + (size_t)ncol * DCH + kb * 32 + koff;
    Q2 q = { *(const uint4*)bp, *(const uint4*)(bp + 16) };
    bfr[kb] = __builtin_bit_cast(v16bf, q);
  }

  for (int s = 0; s < nstripe; ++s) {
    const size_t row0 = ((size_t)blockIdx.x * nstripe + s) * 16;
    __syncthreads();                        // all waves done reading previous tile
    // stage A tile: 512 threads x 16B = 8KB
    const __bf16* gsrc = Ab + row0 * DCH + (size_t)tid * 8;
#if USE_ASYNC_LDS
    {
      unsigned ldsoff = (unsigned)(size_t)(const void*)As + (unsigned)tid * 16u;
      asm volatile("global_load_async_to_lds_b128 %0, %1, off"
                   :: "v"(ldsoff), "v"((const void*)gsrc) : "memory");
      asm volatile("s_wait_asynccnt 0x0" ::: "memory");
    }
#else
    *(uint4*)(As + (size_t)tid * 8) = *(const uint4*)gsrc;
#endif
    __syncthreads();

    v8f acc{};
#pragma unroll
    for (int kb = 0; kb < 8; ++kb) {
      const __bf16* ap = &As[mrow * DCH + kb * 32 + koff];
      Q2 q = { *(const uint4*)ap, *(const uint4*)(ap + 16) };
      v16bf av = __builtin_bit_cast(v16bf, q);
      acc = __builtin_amdgcn_wmma_f32_16x16x32_bf16(
          /*neg_a=*/false, av, /*neg_b=*/false, bfr[kb],
          /*c_mod=*/(short)0, acc, /*reuse_a=*/false, /*reuse_b=*/false);
    }
    const size_t mbase = row0 + (size_t)((lane >> 4) << 3);
#pragma unroll
    for (int v = 0; v < 8; ++v)
      C[(mbase + v) * DCH + ncol] = acc[v];
  }
}

// out[dst] += norm_e * xw[src] ; one thread per (edge, 4 channels)
__global__ __launch_bounds__(256) void k_scatter(const float* __restrict__ xw,
    const int* __restrict__ src, const int* __restrict__ dst,
    const float* __restrict__ norm, float* __restrict__ out, int E) {
  size_t gid = (size_t)blockIdx.x * blockDim.x + threadIdx.x;
  int e = (int)(gid >> 6);
  if (e >= E) return;
  int c = (int)(gid & 63) << 2;
  float w = norm[e];
  float4 x = *(const float4*)(xw + (size_t)src[e] * DCH + c);
  float* o = out + (size_t)dst[e] * DCH + c;
  if (e + 1 < E)
    __builtin_prefetch(xw + (size_t)src[e + 1] * DCH + c, 0, 0);  // global_prefetch_b8
  atomicAdd(o + 0, w * x.x);
  atomicAdd(o + 1, w * x.y);
  atomicAdd(o + 2, w * x.z);
  atomicAdd(o + 3, w * x.w);
}

// h = relu(layernorm(h + dinv^2 * xw)) ; one wave32 per row
__global__ __launch_bounds__(256) void k_selfln(float* __restrict__ h,
    const float* __restrict__ xw, const float* __restrict__ dinv,
    const float* __restrict__ g, const float* __restrict__ b, int N) {
  int row = blockIdx.x * 8 + (threadIdx.x >> 5);
  if (row >= N) return;
  int lane = threadIdx.x & 31;
  float di = dinv[row];
  float d2 = di * di;
  float* hp = h + (size_t)row * DCH + lane * 8;
  const float* xp = xw + (size_t)row * DCH + lane * 8;
  float v[8];
  float s = 0.f, s2 = 0.f;
#pragma unroll
  for (int i = 0; i < 8; ++i) {
    v[i] = hp[i] + d2 * xp[i];
    s += v[i]; s2 += v[i] * v[i];
  }
#pragma unroll
  for (int m = 16; m >= 1; m >>= 1) {       // wave32 reduction
    s  += __shfl_xor(s,  m, 32);
    s2 += __shfl_xor(s2, m, 32);
  }
  float mu  = s * (1.f / DCH);
  float var = s2 * (1.f / DCH) - mu * mu;
  float inv = rsqrtf(var + 1e-5f);
#pragma unroll
  for (int i = 0; i < 8; ++i) {
    float y = (v[i] - mu) * inv * g[lane * 8 + i] + b[lane * 8 + i];
    hp[i] = fmaxf(y, 0.f);
  }
}

// out += dinv^2 * xw  (layer-1 self-loop term; no norm/activation)
__global__ void k_selfadd(float* __restrict__ out, const float* __restrict__ xw,
                          const float* __restrict__ dinv, size_t total) {
  size_t i = (size_t)blockIdx.x * blockDim.x + threadIdx.x;
  if (i >= total) return;
  float di = dinv[i >> 8];
  out[i] += di * di * xw[i];
}

extern "C" void kernel_launch(void* const* d_in, const int* in_sizes, int n_in,
                              void* d_out, int out_size, void* d_ws, size_t ws_size,
                              hipStream_t stream) {
  const int N = in_sizes[0] / DCH;
  const int E = in_sizes[1] / 2;

  const float* X    = (const float*)d_in[0];
  const int*   ei   = (const int*)d_in[1];
  const int*   srcI = ei;        // edge_index[0]
  const int*   dstI = ei + E;    // edge_index[1]
  const float* iw0  = (const float*)d_in[2];
  const float* wih0 = (const float*)d_in[3];
  const float* whh0 = (const float*)d_in[4];
  const float* bih0 = (const float*)d_in[5];
  const float* bhh0 = (const float*)d_in[6];
  const float* lng  = (const float*)d_in[7];
  const float* lnb  = (const float*)d_in[8];
  const float* iw1  = (const float*)d_in[9];
  const float* wih1 = (const float*)d_in[10];
  const float* whh1 = (const float*)d_in[11];
  const float* bih1 = (const float*)d_in[12];
  const float* bhh1 = (const float*)d_in[13];

  // workspace carve-up (~160 MB); d_out doubles as the layer-0 hidden buffer
  char* w = (char*)d_ws;
  auto carve = [&](size_t bytes) { char* p = w; w += (bytes + 255) & ~(size_t)255; return p; };
  const size_t ND = (size_t)N * DCH;
  float*  xw   = (float*)carve(ND * sizeof(float));
  __bf16* Xb   = (__bf16*)carve(ND * sizeof(__bf16));           // bf16 GEMM operand
  float*  dinv = (float*)carve((size_t)N * sizeof(float));      // deg then dinv in-place
  float*  norm = (float*)carve((size_t)E * sizeof(float));
  __bf16* Wt0  = (__bf16*)carve((size_t)DCH * DCH * sizeof(__bf16));
  __bf16* Wt1  = (__bf16*)carve((size_t)DCH * DCH * sizeof(__bf16));
  float*  h    = (float*)d_out;

  dim3 b256(256);
  const unsigned gemmGrid = (unsigned)((N / 16) / STRIPES);     // 100000/16/5 = 1250
  const unsigned cvtGrid  = (unsigned)((ND / 8 + 255) / 256);

  // ---- norm precompute (shared by both layers) ----
  k_fill<<<(N + 255) / 256, b256, 0, stream>>>(dinv, (size_t)N, 1.0f);   // self-loop
  k_deg <<<(E + 255) / 256, b256, 0, stream>>>(dstI, dinv, E);
  k_dinv<<<(N + 255) / 256, b256, 0, stream>>>(dinv, N);
  k_norm<<<(E + 255) / 256, b256, 0, stream>>>(srcI, dstI, dinv, norm, E);
  // ---- GRU weight evolution (both layers, independent) ----
  k_gruw<<<DCH, DCH, 0, stream>>>(iw0, wih0, whh0, bih0, bhh0, Wt0);
  k_gruw<<<DCH, DCH, 0, stream>>>(iw1, wih1, whh1, bih1, bhh1, Wt1);

  // ---- layer 0 ----
  k_cvt <<<cvtGrid, b256, 0, stream>>>(X, Xb, ND / 8);
  k_gemm<<<gemmGrid, 512, 0, stream>>>(Xb, Wt0, xw, STRIPES);
  k_fill<<<(unsigned)((ND + 255) / 256), b256, 0, stream>>>(h, ND, 0.f);
  k_scatter<<<(unsigned)(((size_t)E * 64 + 255) / 256), b256, 0, stream>>>(xw, srcI, dstI, norm, h, E);
  k_selfln<<<(N + 7) / 8, b256, 0, stream>>>(h, xw, dinv, lng, lnb, N);

  // ---- layer 1 (no norm / activation) ----
  k_cvt <<<cvtGrid, b256, 0, stream>>>(h, Xb, ND / 8);
  k_gemm<<<gemmGrid, 512, 0, stream>>>(Xb, Wt1, xw, STRIPES);
  k_fill<<<(unsigned)((ND + 255) / 256), b256, 0, stream>>>(h, ND, 0.f);
  k_scatter<<<(unsigned)(((size_t)E * 64 + 255) / 256), b256, 0, stream>>>(xw, srcI, dstI, norm, h, E);
  k_selfadd<<<(unsigned)((ND + 255) / 256), b256, 0, stream>>>(h, xw, dinv, ND);
}
